// EmotionalTransformer_33088428049068
// MI455X (gfx1250) — compile-verified, matrix operand order checked by reading, not run on hardware
//
#include <hip/hip_runtime.h>

// ---------------------------------------------------------------- constants
constexpr int Bc = 4, Sc = 512, Ec = 768, Hc = 12, Dc = 64, Ic = 3072, Lc = 12, Vc = 50257;
constexpr int TB = 256;

typedef __attribute__((ext_vector_type(16))) __bf16 v16bf;
typedef __attribute__((ext_vector_type(8)))  __bf16 v8bf;
typedef __attribute__((ext_vector_type(8)))  float  v8f;

// Load one 16-element bf16 WMMA fragment: 8 contiguous bf16 at p (K k..k+7 for this lane's
// half) and 8 more at p+16 (K k+16..k+23), per the CDNA5 16-bit A/B VGPR layout.
__device__ inline v16bf load_frag(const __bf16* __restrict__ p)
{
    v8bf lo = *(const v8bf*)(p);
    v8bf hi = *(const v8bf*)(p + 16);
    v16bf f;
#pragma unroll
    for (int i = 0; i < 8; ++i) { f[i] = lo[i]; f[i + 8] = hi[i]; }
    return f;
}

// ---------------------------------------------------------------- WMMA GEMM
// C[z][m][n] = sum_k A[z][m][k] * Bt[z][n][k]  (+bias[n]) (gelu) (+resid) -> f32 or bf16
// A: bf16 row-major [M, ldA]; Bt: bf16 row-major [N, ldB] (i.e. B transposed).
// Each wave computes a 16x64 strip: 1 A-frag reused over 4 B-frags per K-step of 32.
// Out-of-range columns are CLAMPED for loads (results discarded at store time), so the
// K-loop has no divergent control flow and EXEC stays all-ones throughout. The K-loop is
// unrolled 2x so the scheduler overlaps iteration k+1 loads with iteration k WMMAs using
// fresh registers (no rotation copies, no WAR hazard nops).
template<bool BIAS, bool RESID, bool GELU_ACT, bool OUT_BF16>
__global__ __launch_bounds__(TB)
void wmma_gemm_kernel(const __bf16* __restrict__ A, const __bf16* __restrict__ Bt,
                      const float* __restrict__ bias, const float* __restrict__ resid,
                      void* __restrict__ Cout,
                      int M, int N, int K, int ldA, int ldB, int ldC,
                      long long strideA, long long strideB, long long strideC)
{
    const int z = blockIdx.z;
    A  += (size_t)z * strideA;
    Bt += (size_t)z * strideB;

    const int lane = threadIdx.x & 31;
    const int wave = threadIdx.x >> 5;
    const int tiles_n = (N + 63) >> 6;
    const int tiles_m = M >> 4;
    const int tile = blockIdx.x * 8 + wave;
    if (tile >= tiles_m * tiles_n) return;          // wave-uniform exit
    const int m0 = (tile / tiles_n) << 4;
    const int n0 = (tile % tiles_n) << 6;

    const int arow  = m0 + (lane & 15);
    const int khalf = (lane >> 4) << 3;             // lanes 0-15: K 0-7/16-23, lanes 16-31: K 8-15/24-31
    const __bf16* aptr = A + (size_t)arow * ldA + khalf;

    // B column pointers, clamped into range (edge-tile garbage is masked at store time)
    const __bf16* bptr0;
    const __bf16* bptr1;
    const __bf16* bptr2;
    const __bf16* bptr3;
    {
        int c0 = n0 + 0  + (lane & 15); if (c0 > N - 1) c0 = N - 1;
        int c1 = n0 + 16 + (lane & 15); if (c1 > N - 1) c1 = N - 1;
        int c2 = n0 + 32 + (lane & 15); if (c2 > N - 1) c2 = N - 1;
        int c3 = n0 + 48 + (lane & 15); if (c3 > N - 1) c3 = N - 1;
        bptr0 = Bt + (size_t)c0 * ldB + khalf;
        bptr1 = Bt + (size_t)c1 * ldB + khalf;
        bptr2 = Bt + (size_t)c2 * ldB + khalf;
        bptr3 = Bt + (size_t)c3 * ldB + khalf;
    }

    v8f acc[4];
#pragma unroll
    for (int j = 0; j < 4; ++j)
#pragma unroll
        for (int r = 0; r < 8; ++r) acc[j][r] = 0.0f;

#pragma unroll 2
    for (int k0 = 0; k0 < K; k0 += 32) {
        __builtin_prefetch(aptr + k0 + 64, 0, 3);
        v16bf a  = load_frag(aptr  + k0);
        v16bf b0 = load_frag(bptr0 + k0);
        v16bf b1 = load_frag(bptr1 + k0);
        v16bf b2 = load_frag(bptr2 + k0);
        v16bf b3 = load_frag(bptr3 + k0);
        acc[0] = __builtin_amdgcn_wmma_f32_16x16x32_bf16(false, a, false, b0, (short)0, acc[0], false, false);
        acc[1] = __builtin_amdgcn_wmma_f32_16x16x32_bf16(false, a, false, b1, (short)0, acc[1], false, false);
        acc[2] = __builtin_amdgcn_wmma_f32_16x16x32_bf16(false, a, false, b2, (short)0, acc[2], false, false);
        acc[3] = __builtin_amdgcn_wmma_f32_16x16x32_bf16(false, a, false, b3, (short)0, acc[3], false, false);
    }

    float*  cf = (float*)Cout;
    __bf16* cb = (__bf16*)Cout;
    const size_t cbase = (size_t)z * strideC;
#pragma unroll
    for (int r = 0; r < 8; ++r) {
        const int m = m0 + r + ((lane >> 4) << 3);  // D layout: VGPR r -> M=r (+8 for hi lanes)
#pragma unroll
        for (int j = 0; j < 4; ++j) {
            const int col = n0 + j * 16 + (lane & 15);
            if (col < N) {
                float v = acc[j][r];
                if (BIAS)     v += bias[col];
                if (GELU_ACT) v  = 0.5f * v * (1.0f + erff(v * 0.70710678118f));
                const size_t idx = cbase + (size_t)m * ldC + col;
                if (RESID)    v += resid[idx];
                if (OUT_BF16) cb[idx] = (__bf16)v; else cf[idx] = v;
            }
        }
    }
}

// ---------------------------------------------------------------- LayerNorm (f32 in -> bf16 out)
__global__ __launch_bounds__(TB)
void ln_kernel(const float* __restrict__ x, const float* __restrict__ g,
               const float* __restrict__ b, __bf16* __restrict__ out)
{
    const int row = blockIdx.x;
    const float* xr = x + (size_t)row * Ec;
    __shared__ float red[TB];

    float s = 0.0f;
    for (int i = threadIdx.x; i < Ec; i += TB) s += xr[i];
    red[threadIdx.x] = s; __syncthreads();
    for (int o = TB / 2; o > 0; o >>= 1) {
        if (threadIdx.x < o) red[threadIdx.x] += red[threadIdx.x + o];
        __syncthreads();
    }
    const float mean = red[0] * (1.0f / Ec);
    __syncthreads();

    float v = 0.0f;
    for (int i = threadIdx.x; i < Ec; i += TB) { float d = xr[i] - mean; v += d * d; }
    red[threadIdx.x] = v; __syncthreads();
    for (int o = TB / 2; o > 0; o >>= 1) {
        if (threadIdx.x < o) red[threadIdx.x] += red[threadIdx.x + o];
        __syncthreads();
    }
    const float rstd = rsqrtf(red[0] * (1.0f / Ec) + 1e-5f);

    for (int i = threadIdx.x; i < Ec; i += TB)
        out[(size_t)row * Ec + i] = (__bf16)((xr[i] - mean) * rstd * g[i] + b[i]);
}

// ---------------------------------------------------------------- Softmax over S (with scale/bias/mask)
__global__ __launch_bounds__(TB)
void softmax_kernel(const float* __restrict__ scores, const float* __restrict__ ebias,
                    const float* __restrict__ mask, __bf16* __restrict__ probs)
{
    const int row = blockIdx.x;                 // (b*H + h)*S + s
    const int bh = row / Sc;
    const int h_ = bh % Hc;
    const int b_ = bh / Hc;
    const float* sr = scores + (size_t)row * Sc;
    const float* mr = mask + (size_t)b_ * Sc;
    const float eb = ebias[h_];
    __shared__ float red[TB];

    float vals[2];
    float mx = -1e30f;
#pragma unroll
    for (int j = 0; j < 2; ++j) {
        const int i = threadIdx.x + j * TB;
        const float t = sr[i] * 0.125f + eb + mr[i];
        vals[j] = t; mx = fmaxf(mx, t);
    }
    red[threadIdx.x] = mx; __syncthreads();
    for (int o = TB / 2; o > 0; o >>= 1) {
        if (threadIdx.x < o) red[threadIdx.x] = fmaxf(red[threadIdx.x], red[threadIdx.x + o]);
        __syncthreads();
    }
    const float rowmax = red[0];
    __syncthreads();

    float sum = 0.0f;
#pragma unroll
    for (int j = 0; j < 2; ++j) { vals[j] = __expf(vals[j] - rowmax); sum += vals[j]; }
    red[threadIdx.x] = sum; __syncthreads();
    for (int o = TB / 2; o > 0; o >>= 1) {
        if (threadIdx.x < o) red[threadIdx.x] += red[threadIdx.x + o];
        __syncthreads();
    }
    const float inv = 1.0f / red[0];
#pragma unroll
    for (int j = 0; j < 2; ++j)
        probs[(size_t)row * Sc + threadIdx.x + j * TB] = (__bf16)(vals[j] * inv);
}

// ---------------------------------------------------------------- conditioning bias + embedding
__global__ __launch_bounds__(TB)
void prep_bias_kernel(const int* __restrict__ emo_ids, const int* __restrict__ ctx_ids,
                      const int* __restrict__ pers_ids,
                      const float* __restrict__ emo_emb, const float* __restrict__ ctx_emb,
                      const float* __restrict__ pers_emb,
                      const float* __restrict__ W_emo, const float* __restrict__ b_emo,
                      const float* __restrict__ W_ctx, const float* __restrict__ b_ctx,
                      const float* __restrict__ W_pers, const float* __restrict__ b_pers,
                      float* __restrict__ comb)
{
    const int tid = blockIdx.x * TB + threadIdx.x;
    if (tid >= Bc * Ec) return;
    const int e = tid % Ec, b_ = tid / Ec;
    float acc = b_emo[e] + b_ctx[e] + b_pers[e];
    const float* ee = emo_emb + (size_t)emo_ids[b_] * 64;
    for (int i = 0; i < 64; ++i)  acc += ee[i] * W_emo[(size_t)i * Ec + e];
    const float* ce = ctx_emb + (size_t)ctx_ids[b_] * 128;
    for (int i = 0; i < 128; ++i) acc += ce[i] * W_ctx[(size_t)i * Ec + e];
    const float* pe = pers_emb + (size_t)pers_ids[b_] * 96;
    for (int i = 0; i < 96; ++i)  acc += pe[i] * W_pers[(size_t)i * Ec + e];
    comb[tid] = acc;
}

__global__ __launch_bounds__(TB)
void embed_kernel(const int* __restrict__ ids, const int* __restrict__ pos,
                  const float* __restrict__ wte, const float* __restrict__ wpe,
                  const float* __restrict__ comb, float* __restrict__ h)
{
    const size_t tid = (size_t)blockIdx.x * TB + threadIdx.x;
    if (tid >= (size_t)Bc * Sc * Ec) return;
    const int e = tid % Ec;
    const size_t bs = tid / Ec;
    const int b_ = (int)(bs / Sc);
    h[tid] = wte[(size_t)ids[bs] * Ec + e] + wpe[(size_t)pos[bs] * Ec + e] + comb[(size_t)b_ * Ec + e];
}

// ---------------------------------------------------------------- weight prep
// scale[k] = g[k] / ||v[k,:]||   (weight_norm over out-dim per input row)
__global__ __launch_bounds__(TB)
void rowscale_kernel(const float* __restrict__ v, const float* __restrict__ g,
                     float* __restrict__ scale, int K_, int N_)
{
    const int k = blockIdx.x * TB + threadIdx.x;
    if (k >= K_) return;
    const float* row = v + (size_t)k * N_;
    float s = 0.0f;
    for (int n = 0; n < N_; ++n) s += row[n] * row[n];
    scale[k] = g[k] * rsqrtf(s);
}

// in f32 [K_,N_] -> out bf16 [N_,K_] (transposed), optional per-input-row scale
__global__ __launch_bounds__(TB)
void convertT_kernel(const float* __restrict__ in, __bf16* __restrict__ out,
                     int K_, int N_, const float* __restrict__ scale)
{
    const size_t tid = (size_t)blockIdx.x * TB + threadIdx.x;
    const size_t total = (size_t)K_ * N_;
    if (tid >= total) return;
    const int n = (int)(tid % N_);
    const int k = (int)(tid / N_);
    float v = in[tid];
    if (scale) v *= scale[k];
    out[(size_t)n * K_ + k] = (__bf16)v;
}

// ---------------------------------------------------------------- attention data movement
// qkv f32 [B*S, 3E] -> q,k bf16 [B,H,S,D], vT bf16 [B,H,D,S]
__global__ __launch_bounds__(TB)
void qkv_split_kernel(const float* __restrict__ qkv, __bf16* __restrict__ q,
                      __bf16* __restrict__ k, __bf16* __restrict__ vt)
{
    const size_t tid = (size_t)blockIdx.x * TB + threadIdx.x;
    if (tid >= (size_t)Bc * Sc * Ec) return;
    const int e = tid % Ec;
    const size_t bs = tid / Ec;
    const int s_ = (int)(bs % Sc);
    const int b_ = (int)(bs / Sc);
    const int h_ = e / Dc, d_ = e % Dc;
    const float* base = qkv + bs * (3 * Ec);
    const size_t qi = ((((size_t)b_ * Hc + h_) * Sc) + s_) * Dc + d_;
    q[qi] = (__bf16)base[e];
    k[qi] = (__bf16)base[Ec + e];
    const size_t vi = ((((size_t)b_ * Hc + h_) * Dc) + d_) * Sc + s_;
    vt[vi] = (__bf16)base[2 * Ec + e];
}

// o bf16 [B,H,S,D] -> [B,S,E]
__global__ __launch_bounds__(TB)
void opack_kernel(const __bf16* __restrict__ ot, __bf16* __restrict__ o)
{
    const size_t tid = (size_t)blockIdx.x * TB + threadIdx.x;
    if (tid >= (size_t)Bc * Sc * Ec) return;
    const int e = tid % Ec;
    const size_t bs = tid / Ec;
    const int s_ = (int)(bs % Sc);
    const int b_ = (int)(bs / Sc);
    const int h_ = e / Dc, d_ = e % Dc;
    o[tid] = ot[((((size_t)b_ * Hc + h_) * Sc) + s_) * Dc + d_];
}

// ---------------------------------------------------------------- host side
template<bool BIAS, bool RESID, bool GELU_ACT, bool OUT_BF16>
static void launch_gemm(hipStream_t st, const __bf16* A, const __bf16* Bt, const float* bias,
                        const float* resid, void* C, int M, int N, int K,
                        int ldA, int ldB, int ldC,
                        long long sA, long long sB, long long sC, int batches)
{
    const int tiles = (M >> 4) * ((N + 63) >> 6);
    dim3 grid((tiles + 7) / 8, 1, batches);
    wmma_gemm_kernel<BIAS, RESID, GELU_ACT, OUT_BF16><<<grid, TB, 0, st>>>(
        A, Bt, bias, resid, C, M, N, K, ldA, ldB, ldC, sA, sB, sC);
}

static inline size_t rup(size_t x) { return (x + 255) & ~(size_t)255; }

extern "C" void kernel_launch(void* const* d_in, const int* in_sizes, int n_in,
                              void* d_out, int out_size, void* d_ws, size_t ws_size,
                              hipStream_t stream)
{
    (void)in_sizes; (void)n_in; (void)out_size; (void)ws_size;
    // ----- inputs (setup_inputs order)
    const int*   input_ids  = (const int*)  d_in[0];
    const int*   pos_ids    = (const int*)  d_in[1];
    const int*   emo_ids    = (const int*)  d_in[2];
    const int*   ctx_ids    = (const int*)  d_in[3];
    const int*   pers_ids   = (const int*)  d_in[4];
    const float* attn_mask  = (const float*)d_in[5];
    const float* wte        = (const float*)d_in[6];
    const float* wpe        = (const float*)d_in[7];
    const float* emo_emb    = (const float*)d_in[8];
    const float* ctx_emb    = (const float*)d_in[9];
    const float* pers_emb   = (const float*)d_in[10];
    const float* W_emo      = (const float*)d_in[11];
    const float* b_emo      = (const float*)d_in[12];
    const float* W_ctx      = (const float*)d_in[13];
    const float* b_ctx      = (const float*)d_in[14];
    const float* W_pers     = (const float*)d_in[15];
    const float* b_pers     = (const float*)d_in[16];
    const float* ln1_g      = (const float*)d_in[17];
    const float* ln1_b      = (const float*)d_in[18];
    const float* W_qkv      = (const float*)d_in[19];
    const float* b_qkv      = (const float*)d_in[20];
    const float* attn_v     = (const float*)d_in[21];
    const float* attn_g     = (const float*)d_in[22];
    const float* attn_b     = (const float*)d_in[23];
    const float* emo_bias   = (const float*)d_in[24];
    const float* ln2_g      = (const float*)d_in[25];
    const float* ln2_b      = (const float*)d_in[26];
    const float* W_fc       = (const float*)d_in[27];
    const float* b_fc       = (const float*)d_in[28];
    const float* mlp_v      = (const float*)d_in[29];
    const float* mlp_g      = (const float*)d_in[30];
    const float* mlp_b      = (const float*)d_in[31];
    const float* ln_f_g     = (const float*)d_in[32];
    const float* ln_f_b     = (const float*)d_in[33];
    const float* W_lm       = (const float*)d_in[34];

    // ----- workspace carve-out
    char* p = (char*)d_ws;
    auto alloc = [&](size_t bytes) { void* r = p; p += rup(bytes); return r; };
    const size_t T = (size_t)Bc * Sc;                 // 2048 tokens

    float*  comb   = (float*) alloc((size_t)Bc * Ec * 4);
    float*  h      = (float*) alloc(T * Ec * 4);
    __bf16* xb     = (__bf16*)alloc(T * Ec * 2);
    float*  qkv    = (float*) alloc(T * 3 * Ec * 4);
    __bf16* qb     = (__bf16*)alloc((size_t)Bc * Hc * Sc * Dc * 2);
    __bf16* kb     = (__bf16*)alloc((size_t)Bc * Hc * Sc * Dc * 2);
    __bf16* vtb    = (__bf16*)alloc((size_t)Bc * Hc * Dc * Sc * 2);
    float*  scores = (float*) alloc((size_t)Bc * Hc * Sc * Sc * 4);
    __bf16* probs  = (__bf16*)alloc((size_t)Bc * Hc * Sc * Sc * 2);
    __bf16* otmp   = (__bf16*)alloc((size_t)Bc * Hc * Sc * Dc * 2);
    __bf16* ob     = (__bf16*)alloc(T * Ec * 2);
    __bf16* act    = (__bf16*)alloc(T * Ic * 2);
    __bf16* wqkvT  = (__bf16*)alloc((size_t)3 * Ec * Ec * 2);
    __bf16* attnT  = (__bf16*)alloc((size_t)Ec * Ec * 2);
    __bf16* fcT    = (__bf16*)alloc((size_t)Ic * Ec * 2);
    __bf16* mlpT   = (__bf16*)alloc((size_t)Ec * Ic * 2);
    float*  scaleA = (float*) alloc((size_t)Ec * 4);
    float*  scaleM = (float*) alloc((size_t)Ic * 4);
    __bf16* lmT    = (__bf16*)alloc((size_t)Vc * Ec * 2);

    auto blocks = [](size_t n) { return (unsigned)((n + TB - 1) / TB); };

    // ----- embedding + conditioning
    prep_bias_kernel<<<blocks((size_t)Bc * Ec), TB, 0, stream>>>(
        emo_ids, ctx_ids, pers_ids, emo_emb, ctx_emb, pers_emb,
        W_emo, b_emo, W_ctx, b_ctx, W_pers, b_pers, comb);
    embed_kernel<<<blocks(T * Ec), TB, 0, stream>>>(input_ids, pos_ids, wte, wpe, comb, h);

    // ----- LM head weight: f32 [E,V] -> bf16 [V,E]
    convertT_kernel<<<blocks((size_t)Ec * Vc), TB, 0, stream>>>(W_lm, lmT, Ec, Vc, nullptr);

    // ----- transformer blocks
    for (int l = 0; l < Lc; ++l) {
        const float* Wq_l  = W_qkv  + (size_t)l * Ec * 3 * Ec;
        const float* av_l  = attn_v + (size_t)l * Ec * Ec;
        const float* fc_l  = W_fc   + (size_t)l * Ec * Ic;
        const float* mv_l  = mlp_v  + (size_t)l * Ic * Ec;

        convertT_kernel<<<blocks((size_t)Ec * 3 * Ec), TB, 0, stream>>>(Wq_l, wqkvT, Ec, 3 * Ec, nullptr);
        rowscale_kernel<<<blocks(Ec), TB, 0, stream>>>(av_l, attn_g + (size_t)l * Ec, scaleA, Ec, Ec);
        convertT_kernel<<<blocks((size_t)Ec * Ec), TB, 0, stream>>>(av_l, attnT, Ec, Ec, scaleA);
        convertT_kernel<<<blocks((size_t)Ec * Ic), TB, 0, stream>>>(fc_l, fcT, Ec, Ic, nullptr);
        rowscale_kernel<<<blocks(Ic), TB, 0, stream>>>(mv_l, mlp_g + (size_t)l * Ic, scaleM, Ic, Ec);
        convertT_kernel<<<blocks((size_t)Ic * Ec), TB, 0, stream>>>(mv_l, mlpT, Ic, Ec, scaleM);

        // x = LN1(h) ; qkv = x @ Wq + bq
        ln_kernel<<<(unsigned)T, TB, 0, stream>>>(h, ln1_g + (size_t)l * Ec, ln1_b + (size_t)l * Ec, xb);
        launch_gemm<true, false, false, false>(stream, xb, wqkvT, b_qkv + (size_t)l * 3 * Ec, nullptr,
            qkv, (int)T, 3 * Ec, Ec, Ec, Ec, 3 * Ec, 0, 0, 0, 1);
        qkv_split_kernel<<<blocks(T * Ec), TB, 0, stream>>>(qkv, qb, kb, vtb);

        // scores[b,h] = q @ k^T  (batched over B*H)
        launch_gemm<false, false, false, false>(stream, qb, kb, nullptr, nullptr, scores,
            Sc, Sc, Dc, Dc, Dc, Sc,
            (long long)Sc * Dc, (long long)Sc * Dc, (long long)Sc * Sc, Bc * Hc);
        softmax_kernel<<<(unsigned)((size_t)Bc * Hc * Sc), TB, 0, stream>>>(
            scores, emo_bias + (size_t)l * Hc, attn_mask, probs);

        // o[b,h] = probs @ v  (v stored transposed for contiguous-K WMMA)
        launch_gemm<false, false, false, true>(stream, probs, vtb, nullptr, nullptr, otmp,
            Sc, Dc, Sc, Sc, Sc, Dc,
            (long long)Sc * Sc, (long long)Dc * Sc, (long long)Sc * Dc, Bc * Hc);
        opack_kernel<<<blocks(T * Ec), TB, 0, stream>>>(otmp, ob);

        // h = h + o @ wn(attn_v, attn_g) + attn_b
        launch_gemm<true, true, false, false>(stream, ob, attnT, attn_b + (size_t)l * Ec, h,
            h, (int)T, Ec, Ec, Ec, Ec, Ec, 0, 0, 0, 1);

        // act = gelu(LN2(h) @ W_fc + b_fc)   (GELU + bf16 store fused into GEMM epilogue)
        ln_kernel<<<(unsigned)T, TB, 0, stream>>>(h, ln2_g + (size_t)l * Ec, ln2_b + (size_t)l * Ec, xb);
        launch_gemm<true, false, true, true>(stream, xb, fcT, b_fc + (size_t)l * Ic, nullptr,
            act, (int)T, Ic, Ec, Ec, Ec, Ic, 0, 0, 0, 1);

        // h = h + act @ wn(mlp_v, mlp_g) + mlp_b
        launch_gemm<true, true, false, false>(stream, act, mlpT, mlp_b + (size_t)l * Ec, h,
            h, (int)T, Ec, Ic, Ic, Ic, Ec, 0, 0, 0, 1);
    }

    // ----- final LN + LM head
    ln_kernel<<<(unsigned)T, TB, 0, stream>>>(h, ln_f_g, ln_f_b, xb);
    launch_gemm<false, false, false, false>(stream, xb, lmT, nullptr, nullptr, d_out,
        (int)T, Vc, Ec, Ec, Ec, Vc, 0, 0, 0, 1);
}
